// additiveAttention_48034914238887
// MI455X (gfx1250) — matrix operand order repaired and checked by reading, not compiled, not run on hardware
//
#include <hip/hip_runtime.h>

// ---------------------------------------------------------------------------
// Fused additive attention for MI455X (gfx1250, wave32, WMMA bf16).
//   energy = tanh([q||k] @ [Wq||Wk]^T + (bq+bk)) + mask
//   out    = softmax(energy) @ v
// One workgroup (16 waves, 512 thr) per (batch, 32-row s-tile).
// tanh bounds the logits in [-1,1] (+mask), so softmax needs no max shift:
// denominator is a per-lane running sum reduced once at the end. This keeps
// the register live-set small (no spills) and needs only 2 barriers/chunk.
// All GEMMs via v_wmma_f32_16x16x32_bf16; weight / v B-fragments each feed
// two WMMAs (two 16-row blocks), halving L2 traffic per FLOP.
// ---------------------------------------------------------------------------

typedef __bf16 bf16_t;
typedef bf16_t v16bf __attribute__((ext_vector_type(16)));
typedef float  v8f   __attribute__((ext_vector_type(8)));

#define B_   8
#define S_   2048
#define E_   1024
#define C_   2048

#define ROWS_   32            // s-rows per workgroup (2 row-blocks of 16)
#define NWAVE   16            // waves per workgroup (512 threads)
#define CW_     256           // C columns per chunk = NWAVE * 16
#define NCHUNK  (C_ / CW_)    // 8
#define NT_     4             // 16-col n-tiles per wave in P@V (E/NWAVE/16)

#define XLD  2056             // lds_X row stride (bf16): 2048 + 8 pad
#define PLD  264              // lds_P row stride (bf16): 256 + 8 pad

union AFrag { v16bf v; uint4 u[2]; };

__device__ __forceinline__ float fast_tanh(float x) {
    // tanh(x) = 1 - 2/(exp(2x)+1); branch-free, native v_exp/v_rcp.
    const float e = __builtin_amdgcn_exp2f(x * 2.8853900817779268f); // exp(2x)
    return 1.0f - 2.0f * __builtin_amdgcn_rcpf(e + 1.0f);
}
__device__ __forceinline__ float fast_exp(float x) {
    return __builtin_amdgcn_exp2f(x * 1.4426950408889634f);
}

__global__ __launch_bounds__(512)
void additive_attn_fused(const float* __restrict__ q,  const float* __restrict__ k,
                         const float* __restrict__ v,  const float* __restrict__ mask,
                         const float* __restrict__ Wq, const float* __restrict__ bq,
                         const float* __restrict__ Wk, const float* __restrict__ bk,
                         float* __restrict__ out)
{
    __shared__ bf16_t lds_X[ROWS_ * XLD];      // bf16([q_tile || k_tile]), 32 x 2048
    __shared__ bf16_t lds_P[ROWS_ * PLD];      // softmax-numerator chunk, 32 x 256
    __shared__ float  lds_ssum[NWAVE][ROWS_];  // final cross-wave denom reduction

    const int tid   = threadIdx.x;
    const int wave  = tid >> 5;
    const int lane  = tid & 31;
    const int nlane = lane & 15;   // column / row-within-16 index of this lane
    const int khalf = lane >> 4;   // which K-half of the fragment this lane holds
    const int rbase = khalf * 8;   // C/D rows owned by this half-wave

    const int b  = blockIdx.y;
    const int s0 = blockIdx.x * ROWS_;

    // ---- Stage X = bf16([q_tile || k_tile]) into LDS (32 x 2048) ----
    {
        const float* qrow = q + ((size_t)b * S_ + s0) * E_;
        const float* krow = k + ((size_t)b * S_ + s0) * E_;
        for (int idx = tid; idx < ROWS_ * 2048 / 4; idx += 512) {
            const int row  = idx >> 9;            // 512 float4 per row
            const int col4 = (idx & 511) * 4;
            float4 f;
            if (col4 < E_) f = *(const float4*)(qrow + (size_t)row * E_ + col4);
            else           f = *(const float4*)(krow + (size_t)row * E_ + (col4 - E_));
            bf16_t* dst = lds_X + row * XLD + col4;
            dst[0] = (bf16_t)f.x; dst[1] = (bf16_t)f.y;
            dst[2] = (bf16_t)f.z; dst[3] = (bf16_t)f.w;
        }
    }
    __syncthreads();

    // ---- Per-lane running state ----
    float lsum[2][8];                   // per-lane partial softmax denominators
    v8f   o_acc[2][NT_];                // 2 row-blocks x (16x64) f32 output slice
    const v8f vzero = {0.f,0.f,0.f,0.f,0.f,0.f,0.f,0.f};
#pragma unroll
    for (int rt = 0; rt < 2; ++rt) {
#pragma unroll
        for (int i = 0; i < 8; ++i) lsum[rt][i] = 0.f;
#pragma unroll
        for (int nt = 0; nt < NT_; ++nt) o_acc[rt][nt] = vzero;
    }

    // Hoisted bases (32-bit offsets per chunk keep addressing cheap)
    const int   ncol   = wave * 16 + nlane;                 // this lane's column id
    const float* bqp   = bq + ncol;
    const float* bkp   = bk + ncol;
    const float* mbase0 = mask + ((size_t)b * S_ + s0 + 0  + rbase) * C_ + ncol;
    const float* mbase1 = mask + ((size_t)b * S_ + s0 + 16 + rbase) * C_ + ncol;

#pragma unroll 1
    for (int chunk = 0; chunk < NCHUNK; ++chunk) {
        const int c0 = chunk * CW_;
        const int n  = c0 + ncol;                 // energy column this lane owns

        // ---- 1) Energy tiles: two 16x16 row-blocks, shared weight B-frag ----
        v8f e_acc0 = vzero, e_acc1 = vzero;
        const float* wqrow = Wq + (size_t)n * E_;
        const float* wkrow = Wk + (size_t)n * E_;
#pragma unroll 1
        for (int ks = 0; ks < 64; ++ks) {         // K = 2048, 32 per WMMA
            // B: column n of Wcat^T == contiguous row of Wq/Wk
            const float* wrow = (ks < 32) ? (wqrow + ks * 32) : (wkrow + (ks - 32) * 32);
            const float4* wv  = (const float4*)(wrow + khalf * 16);
            float wf[16];
            *(float4*)(wf + 0)  = wv[0];
            *(float4*)(wf + 4)  = wv[1];
            *(float4*)(wf + 8)  = wv[2];
            *(float4*)(wf + 12) = wv[3];
            AFrag bw;
#pragma unroll
            for (int j = 0; j < 16; ++j) bw.v[j] = (bf16_t)wf[j];

            // A frags for both row-blocks (contiguous K-halves, matching B)
            AFrag a0, a1;
            const bf16_t* x0 = lds_X + (0  + nlane) * XLD + ks * 32;
            const bf16_t* x1 = lds_X + (16 + nlane) * XLD + ks * 32;
            a0.u[0] = *(const uint4*)(x0 + khalf * 16);
            a0.u[1] = *(const uint4*)(x0 + khalf * 16 + 8);
            a1.u[0] = *(const uint4*)(x1 + khalf * 16);
            a1.u[1] = *(const uint4*)(x1 + khalf * 16 + 8);

            e_acc0 = __builtin_amdgcn_wmma_f32_16x16x32_bf16(
                         false, a0.v, false, bw.v, (short)0, e_acc0, false, false);
            e_acc1 = __builtin_amdgcn_wmma_f32_16x16x32_bf16(
                         false, a1.v, false, bw.v, (short)0, e_acc1, false, false);
        }

        // ---- 2) p = exp(tanh(e+bias)+mask); write P-tile; accumulate denom ----
        const float bias = bqp[c0] + bkp[c0];
#pragma unroll
        for (int rt = 0; rt < 2; ++rt) {
            const float* mrow = (rt == 0) ? (mbase0 + c0) : (mbase1 + c0);
#pragma unroll
            for (int r = 0; r < 8; ++r) {
                const float e = (rt == 0) ? e_acc0[r] : e_acc1[r];
                const float p = fast_exp(fast_tanh(e + bias) + mrow[(size_t)r * C_]);
                lds_P[(rt * 16 + rbase + r) * PLD + wave * 16 + nlane] = (bf16_t)p;
                lsum[rt][r] += p;
            }
        }
        __syncthreads();   // P-tile visible to all waves

        // ---- 3) O += P(32x256) @ V[c0:c0+256, wave-E-slice]; shared v B-frag ----
        const int e0 = wave * 64;
#pragma unroll 1
        for (int nt = 0; nt < NT_; ++nt) {
            const int ecol = e0 + nt * 16 + nlane;
#pragma unroll
            for (int ks = 0; ks < CW_ / 32; ++ks) {       // K = 256 per chunk
                AFrag bv;
                const float* vp =
                    v + ((size_t)b * C_ + (c0 + ks * 32 + khalf * 16)) * E_ + ecol;
#pragma unroll
                for (int j = 0; j < 16; ++j) bv.v[j] = (bf16_t)vp[(size_t)j * E_];

                AFrag a0, a1;
                const bf16_t* p0 = lds_P + (0  + nlane) * PLD + ks * 32;
                const bf16_t* p1 = lds_P + (16 + nlane) * PLD + ks * 32;
                a0.u[0] = *(const uint4*)(p0 + khalf * 16);
                a0.u[1] = *(const uint4*)(p0 + khalf * 16 + 8);
                a1.u[0] = *(const uint4*)(p1 + khalf * 16);
                a1.u[1] = *(const uint4*)(p1 + khalf * 16 + 8);

                o_acc[0][nt] = __builtin_amdgcn_wmma_f32_16x16x32_bf16(
                                   false, a0.v, false, bv.v, (short)0, o_acc[0][nt], false, false);
                o_acc[1][nt] = __builtin_amdgcn_wmma_f32_16x16x32_bf16(
                                   false, a1.v, false, bv.v, (short)0, o_acc[1][nt], false, false);
            }
        }
        __syncthreads();   // protect lds_P for next chunk
    }

    // ---- Final softmax denominator: lane butterfly + one cross-wave reduce ----
#pragma unroll
    for (int rt = 0; rt < 2; ++rt)
#pragma unroll
        for (int r = 0; r < 8; ++r) {
            float s = lsum[rt][r];
            s += __shfl_xor(s, 1, 32);
            s += __shfl_xor(s, 2, 32);
            s += __shfl_xor(s, 4, 32);
            s += __shfl_xor(s, 8, 32);
            if (nlane == 0) lds_ssum[wave][rt * 16 + rbase + r] = s;
        }
    __syncthreads();

    // ---- Epilogue: normalize and store fp32 output ----
#pragma unroll
    for (int rt = 0; rt < 2; ++rt) {
        float rinv[8];
#pragma unroll
        for (int r = 0; r < 8; ++r) {
            float l = 0.f;
#pragma unroll
            for (int w2 = 0; w2 < NWAVE; ++w2) l += lds_ssum[w2][rt * 16 + rbase + r];
            rinv[r] = __builtin_amdgcn_rcpf(l);
        }
#pragma unroll
        for (int nt = 0; nt < NT_; ++nt) {
            const int ecol = wave * 64 + nt * 16 + nlane;
#pragma unroll
            for (int r = 0; r < 8; ++r) {
                out[((size_t)b * S_ + (s0 + rt * 16 + rbase + r)) * E_ + ecol] =
                    o_acc[rt][nt][r] * rinv[r];
            }
        }
    }
}

extern "C" void kernel_launch(void* const* d_in, const int* in_sizes, int n_in,
                              void* d_out, int out_size, void* d_ws, size_t ws_size,
                              hipStream_t stream) {
    const float* q    = (const float*)d_in[0];
    const float* k    = (const float*)d_in[1];
    const float* v    = (const float*)d_in[2];
    const float* mask = (const float*)d_in[3];
    const float* Wq   = (const float*)d_in[4];
    const float* bq   = (const float*)d_in[5];
    const float* Wk   = (const float*)d_in[6];
    const float* bk   = (const float*)d_in[7];
    float* out = (float*)d_out;

    dim3 grid(S_ / ROWS_, B_, 1);   // 64 s-tiles x 8 batches = 512 workgroups
    additive_attn_fused<<<grid, 512, 0, stream>>>(q, k, v, mask, Wq, bq, Wk, bk, out);
}
